// CTDEncoder_81518479278476
// MI455X (gfx1250) — matrix operand order repaired and checked by reading, not compiled, not run on hardware
//
#include <hip/hip_runtime.h>
#include <hip/hip_bf16.h>

// ---------------------------------------------------------------------------
// Weighted-relational GCN (8 layers) for MI455X / gfx1250, wave32 + WMMA.
//
//   wrgcn(h) = [h | agg0 | agg1 | agg2] @ [root; W0; W1; W2] + bias
//
// Irregular gather/scatter: f32 atomics (HBM bound).
// Dense part: one bf16 WMMA GEMM per layer. Both operands are pre-swizzled
// into the exact v_wmma_f32_16x16x32_bf16 fragment layout in global memory,
// so the GEMM inner loop is pure immediate-offset global_load_b128 -> v_wmma
// (no LDS, no barriers; A-fragments ~100MB sit in the 192MB L2, weights are
// L2-hot). KT and NT are template constants so all fragment addresses fold
// into the 24-bit IOFFSET field.
// ---------------------------------------------------------------------------

#define N_X    40000
#define DIS_N  10000
#define NN     (N_X + DIS_N)    // 50000 nodes
#define MT     (NN / 16)        // 3125 M-tiles (exact: 50000 % 16 == 0)
#define E_CNT  800000
#define BN_EPS 1e-5f

typedef __attribute__((ext_vector_type(16))) __bf16 v16bf;
typedef __attribute__((ext_vector_type(8)))  float  v8f;

static __device__ __forceinline__ unsigned short f2bf(float f) {
  unsigned int u = __float_as_uint(f);
  u += 0x7FFFu + ((u >> 16) & 1u);          // round-to-nearest-even
  return (unsigned short)(u >> 16);
}

// ---------------------------------------------------------------- utilities
__global__ void k_zero(float* __restrict__ p, long n) {
  long i = (long)blockIdx.x * blockDim.x + threadIdx.x;
  if (i < n) p[i] = 0.f;
}

__global__ void k_deg(const int* __restrict__ dst, float* __restrict__ deg, int e) {
  int i = blockIdx.x * blockDim.x + threadIdx.x;
  if (i < e) atomicAdd(&deg[dst[i]], 1.0f);
}

__global__ void k_dis(float* __restrict__ deg, int n) {   // in-place deg -> d^-1/2
  int i = blockIdx.x * blockDim.x + threadIdx.x;
  if (i < n) {
    float d = deg[i];
    deg[i] = (d > 0.f) ? rsqrtf(fmaxf(d, 1.f)) : 0.f;
  }
}

__global__ void k_ew(const int* __restrict__ dst, const int* __restrict__ src,
                     const float* __restrict__ dis, float* __restrict__ ew, int e) {
  int i = blockIdx.x * blockDim.x + threadIdx.x;
  if (i < e) ew[i] = dis[dst[i]] * dis[src[i]];
}

__global__ void k_concat(const float* __restrict__ x, const float* __restrict__ emb,
                         float* __restrict__ h) {
  long i = (long)blockIdx.x * blockDim.x + threadIdx.x;
  const long nx = (long)N_X * 128;
  const long nt = (long)NN * 128;
  if (i < nt) h[i] = (i < nx) ? x[i] : emb[i - nx];
}

// ------------------------------------------------------ gather + scatter-add
// one 128-thread block per edge; coalesced row gather, f32 atomic scatter
__global__ __launch_bounds__(128)
void k_scatter(const float* __restrict__ h, const int* __restrict__ src,
               const int* __restrict__ dst, const int* __restrict__ et,
               const float* __restrict__ ew, float* __restrict__ agg,
               int d_in, int relu) {
  int e = blockIdx.x;
  float w = ew[e];
  if (w == 0.f) return;
  int s = src[e], d = dst[e], r = et[e];
  const float* hs = h + (long)s * d_in;
  float* ag = agg + (long)d * (3 * d_in) + (long)r * d_in;
  __builtin_prefetch(hs, 0, 0);            // global_prefetch_b8
  for (int c = threadIdx.x; c < d_in; c += 128) {
    float v = hs[c];
    if (relu) v = fmaxf(v, 0.f);
    atomicAdd(&ag[c], w * v);
  }
}

// ---------------------------------------------------------------------------
// Pack A = [h | agg0 | agg1 | agg2] directly into WMMA A-fragment layout:
//   Af[((mt*KT + kt)*32 + lane)*16 + e]
//   lane<16:  row = mt*16+lane, K = kt*32 + (e<8 ? e   : 16+e-8)
//   lane>=16: row = mt*16+l16,  K = kt*32 + (e<8 ? 8+e : 24+e-8)
// (ISA 7.12.2, 16-bit A-matrix 16x32)
// ---------------------------------------------------------------------------
__global__ void k_packA_frag(const float* __restrict__ h, const float* __restrict__ agg,
                             unsigned short* __restrict__ Af, int d_in, int relu, int KT) {
  long total = (long)MT * KT * 512;
  long i = (long)blockIdx.x * blockDim.x + threadIdx.x;
  if (i >= total) return;
  int  e    = (int)(i & 15);
  int  lane = (int)((i >> 4) & 31);
  long tile = i >> 9;                 // mt*KT + kt
  int  kt   = (int)(tile % KT);
  long mt   = tile / KT;
  int  half = lane >> 4;
  int  row  = (int)(mt * 16 + (lane & 15));
  int  k    = kt * 32 + ((e < 8) ? (half * 8 + e) : (16 + half * 8 + (e - 8)));
  float v;
  if (k < d_in) {
    v = h[(long)row * d_in + k];
    if (relu) v = fmaxf(v, 0.f);
  } else {
    v = agg[(long)row * (3 * d_in) + (k - d_in)];
  }
  Af[i] = f2bf(v);
}

// ---------------------------------------------------------------------------
// Pack W = [root; W0; W1; W2] into WMMA B-fragment layout:
//   Wf[((kt*NT + nt)*32 + lane)*16 + e],  k = kt*32 + (e<8 ? (lane>>4)*8+e : 16+(lane>>4)*8+e-8),
//   col = nt*16 + (lane&15)      (ISA 7.12.2, 16-bit B-matrix 32x16)
// ---------------------------------------------------------------------------
__global__ void k_packW_frag(const float* __restrict__ root, const float* __restrict__ w3,
                             unsigned short* __restrict__ Wf, int d_in, int d_out) {
  int  KT = (4 * d_in) >> 5;
  int  NT = d_out >> 4;
  long total = (long)KT * NT * 512;
  long i = (long)blockIdx.x * blockDim.x + threadIdx.x;
  if (i >= total) return;
  int  e    = (int)(i & 15);
  int  lane = (int)((i >> 4) & 31);
  long tile = i >> 9;                 // kt*NT + nt
  int  nt   = (int)(tile % NT);
  int  kt   = (int)(tile / NT);
  int  k    = kt * 32 + ((e < 8) ? ((lane >> 4) * 8 + e) : (16 + (lane >> 4) * 8 + (e - 8)));
  int  col  = nt * 16 + (lane & 15);
  float v = (k < d_in) ? root[(long)k * d_out + col]
                       : w3[(long)(k - d_in) * d_out + col];
  Wf[i] = f2bf(v);
}

// ---------------------------------------------------------------------------
// Fragment-direct WMMA GEMM. 256 threads = 8 wave32 waves; wave w owns M-tile
// mt = blockIdx.x*8 + w and a 16x64 output strip (4 accumulators).
// KT = K/32 tiles, NT = Nc/16 tiles are compile-time so every fragment load
// is base + immediate offset (24-bit IOFFSET).
// ---------------------------------------------------------------------------
template <int KT, int NT>
__global__ __launch_bounds__(256)
void k_gemm_frag(const unsigned short* __restrict__ Af,
                 const unsigned short* __restrict__ Wf,
                 const float* __restrict__ bias, float* __restrict__ C) {
  constexpr int Nc = NT * 16;
  const int lane = threadIdx.x & 31;
  const int wave = threadIdx.x >> 5;
  const int mt   = blockIdx.x * 8 + wave;
  if (mt >= MT) return;                      // uniform per wave: EXEC stays all-1
  const int nt0 = blockIdx.y * 4;

  const v8f vzero = {0.f, 0.f, 0.f, 0.f, 0.f, 0.f, 0.f, 0.f};
  v8f acc[4];
#pragma unroll
  for (int t = 0; t < 4; ++t) acc[t] = vzero;

  const unsigned short* ap = Af + ((long)mt * KT) * 512 + lane * 16;
  const unsigned short* bp = Wf + ((long)nt0 * 32 + lane) * 16;
#pragma unroll 8
  for (int kt = 0; kt < KT; ++kt) {
    union { uint4 q[2]; v16bf v; } a;
    a.q[0] = *(const uint4*)(ap + (long)kt * 512);
    a.q[1] = *(const uint4*)(ap + (long)kt * 512 + 8);
    __builtin_prefetch(ap + (long)(kt + 1) * 512, 0, 0);   // next A K-tile
#pragma unroll
    for (int t = 0; t < 4; ++t) {
      union { uint4 q[2]; v16bf v; } b;
      const unsigned short* bpt = bp + ((long)kt * NT + t) * 512;
      b.q[0] = *(const uint4*)(bpt);
      b.q[1] = *(const uint4*)(bpt + 8);
      acc[t] = __builtin_amdgcn_wmma_f32_16x16x32_bf16(
          false, a.v, false, b.v, (short)0, acc[t], false, false);
    }
  }

  const int half = lane >> 4, l16 = lane & 15;
#pragma unroll
  for (int t = 0; t < 4; ++t) {
    int col = (nt0 + t) * 16 + l16;
    float bv = bias[col];
#pragma unroll
    for (int v = 0; v < 8; ++v) {
      int row = mt * 16 + half * 8 + v;      // C/D: VGPR v -> M = half*8+v
      C[(long)row * Nc + col] = acc[t][v] + bv;
    }
  }
}

// ------------------------------------------------------------- batch norm
__global__ __launch_bounds__(256)
void k_bnstats(const float* __restrict__ h, float* __restrict__ mean,
               float* __restrict__ var, int d) {
  __shared__ float s1[256], s2[256];
  int c = blockIdx.x;
  float sum = 0.f, sq = 0.f;
  for (int r = threadIdx.x; r < NN; r += 256) {
    float v = h[(long)r * d + c];
    sum += v; sq += v * v;
  }
  s1[threadIdx.x] = sum; s2[threadIdx.x] = sq;
  __syncthreads();
  for (int s = 128; s > 0; s >>= 1) {
    if (threadIdx.x < s) { s1[threadIdx.x] += s1[threadIdx.x + s];
                           s2[threadIdx.x] += s2[threadIdx.x + s]; }
    __syncthreads();
  }
  if (threadIdx.x == 0) {
    float m = s1[0] / (float)NN;
    mean[c] = m;
    var[c]  = s2[0] / (float)NN - m * m;   // biased, matches jnp.var
  }
}

__global__ void k_bnapply(const float* __restrict__ hin, float* __restrict__ hout,
                          const float* __restrict__ mean, const float* __restrict__ var,
                          const float* __restrict__ g, const float* __restrict__ b, int d) {
  long i = (long)blockIdx.x * blockDim.x + threadIdx.x;
  if (i >= (long)NN * d) return;
  int c = (int)(i % d);
  hout[i] = (hin[i] - mean[c]) * rsqrtf(var[c] + BN_EPS) * g[c] + b[c];
}

// ---------------------------------------------------------------- launcher
extern "C" void kernel_launch(void* const* d_in, const int* in_sizes, int n_in,
                              void* d_out, int out_size, void* d_ws, size_t ws_size,
                              hipStream_t stream) {
  const float* x   = (const float*)d_in[0];
  const float* emb = (const float*)d_in[1];
  const float* cw[4] = {(const float*)d_in[2],  (const float*)d_in[5],
                        (const float*)d_in[8],  (const float*)d_in[11]};
  const float* rt[4] = {(const float*)d_in[3],  (const float*)d_in[6],
                        (const float*)d_in[9],  (const float*)d_in[12]};
  const float* bs[4] = {(const float*)d_in[4],  (const float*)d_in[7],
                        (const float*)d_in[10], (const float*)d_in[13]};
  const float* n1g = (const float*)d_in[14];
  const float* n1b = (const float*)d_in[15];
  const float* n2g = (const float*)d_in[16];
  const float* n2b = (const float*)d_in[17];
  const int* eidx = (const int*)d_in[18];
  const int* etyp = (const int*)d_in[19];
  const int* dst = eidx;            // edge_index[0]
  const int* src = eidx + E_CNT;    // edge_index[1]

  // workspace carve-out (256B aligned). h_out aliases agg (dead after packA).
  char* ws = (char*)d_ws;
  size_t off = 0;
  auto carve = [&](size_t bytes) -> void* {
    off = (off + 255) & ~(size_t)255;
    void* p = ws + off;
    off += bytes;
    return p;
  };
  float*          h_cur = (float*)carve((size_t)NN * 256 * 4);
  float*          agg   = (float*)carve((size_t)NN * 768 * 4);
  float*          h_out = agg;                       // alias (sequenced safely)
  unsigned short* Abuf  = (unsigned short*)carve((size_t)NN * 1024 * 2);
  unsigned short* Wbuf  = (unsigned short*)carve((size_t)1024 * 256 * 2);
  float*          ewb   = (float*)carve((size_t)E_CNT * 4);
  float*          disb  = (float*)carve((size_t)NN * 4);
  float*          meanb = (float*)carve(256 * 4);
  float*          varb  = (float*)carve(256 * 4);
  (void)ws_size; (void)in_sizes; (void)n_in; (void)out_size;

  // ---- gcn_norm edge weights
  k_zero<<<(NN + 255) / 256, 256, 0, stream>>>(disb, NN);
  k_deg<<<(E_CNT + 255) / 256, 256, 0, stream>>>(dst, disb, E_CNT);
  k_dis<<<(NN + 255) / 256, 256, 0, stream>>>(disb, NN);
  k_ew<<<(E_CNT + 255) / 256, 256, 0, stream>>>(dst, src, disb, ewb, E_CNT);
  k_concat<<<((long)NN * 128 + 255) / 256, 256, 0, stream>>>(x, emb, h_cur);

  struct Layer {
    int din, dout;
    const float *w, *rtp, *bsp, *g, *b;
    int relu, bn;
  };
  Layer layers[8] = {
    {128, 128, cw[0], rt[0], bs[0], n1g, n1b, 0, 1},
    {128, 256, cw[1], rt[1], bs[1], n2g, n2b, 1, 1},
    {256, 256, cw[2], rt[2], bs[2], n2g, n2b, 1, 1},
    {256, 256, cw[2], rt[2], bs[2], n2g, n2b, 1, 1},
    {256, 256, cw[2], rt[2], bs[2], n2g, n2b, 1, 1},
    {256, 256, cw[2], rt[2], bs[2], n2g, n2b, 1, 1},
    {256, 256, cw[2], rt[2], bs[2], n2g, n2b, 1, 1},
    {256,  64, cw[3], rt[3], bs[3], nullptr, nullptr, 1, 0},
  };

  for (int li = 0; li < 8; ++li) {
    const Layer& L = layers[li];
    const int K  = 4 * L.din;
    const int KT = K >> 5;              // 16 or 32 K-tiles

    long aggN = (long)NN * 3 * L.din;
    k_zero<<<(aggN + 255) / 256, 256, 0, stream>>>(agg, aggN);
    k_scatter<<<E_CNT, 128, 0, stream>>>(h_cur, src, dst, etyp, ewb, agg,
                                         L.din, L.relu);
    long an = (long)MT * KT * 512;
    k_packA_frag<<<(an + 255) / 256, 256, 0, stream>>>(h_cur, agg, Abuf,
                                                       L.din, L.relu, KT);
    long wn = (long)KT * (L.dout >> 4) * 512;
    k_packW_frag<<<(wn + 255) / 256, 256, 0, stream>>>(L.rtp, L.w, Wbuf,
                                                       L.din, L.dout);

    float* outp = L.bn ? h_out : (float*)d_out;
    dim3 gg((MT + 7) / 8, L.dout / 64);
    if (KT == 16 && L.dout == 128)
      k_gemm_frag<16, 8><<<gg, 256, 0, stream>>>(Abuf, Wbuf, L.bsp, outp);
    else if (KT == 16 && L.dout == 256)
      k_gemm_frag<16, 16><<<gg, 256, 0, stream>>>(Abuf, Wbuf, L.bsp, outp);
    else if (KT == 32 && L.dout == 256)
      k_gemm_frag<32, 16><<<gg, 256, 0, stream>>>(Abuf, Wbuf, L.bsp, outp);
    else
      k_gemm_frag<32, 4><<<gg, 256, 0, stream>>>(Abuf, Wbuf, L.bsp, outp);

    if (L.bn) {
      k_bnstats<<<L.dout, 256, 0, stream>>>(h_out, meanb, varb, L.dout);
      long bn = (long)NN * L.dout;
      k_bnapply<<<(bn + 255) / 256, 256, 0, stream>>>(h_out, h_cur, meanb, varb,
                                                      L.g, L.b, L.dout);
    }
  }
}
